// GaussianSplattingV2_86346022519336
// MI455X (gfx1250) — matrix operand order repaired and checked by reading, not compiled, not run on hardware
//
#include <hip/hip_runtime.h>
#include <hip/hip_bf16.h>

// ---------------- CDNA5 WMMA types ----------------
typedef __attribute__((ext_vector_type(16))) _Float16 v16h;
typedef __attribute__((ext_vector_type(8)))  _Float16 v8h;
typedef __attribute__((ext_vector_type(8)))  float    v8f;

union V16H { v16h v; v8h h[2]; };

// ---------------- problem constants ----------------
#define H_IN  24
#define W_IN  24
#define C_IN  64
#define HIDC  256
#define C4    1024
#define HF    48
#define WF    48
#define NPIX  2304      // 48*48
#define HO    96
#define WO    96
#define PIXO  9216      // 96*96
#define ND    730
#define NDP   736       // 23 * 32

#define ACT_NONE 0
#define ACT_RELU 1
#define ACT_LRELU 2
#define ACT_TANH 3
#define ACT_SIG 4

// ============================================================
// pack f32 [M,K] -> f16 [Mp,Kp] zero-padded (weights + activations)
// ============================================================
__global__ void pack_f16_kernel(const float* __restrict__ in, _Float16* __restrict__ out,
                                int M, int K, int Mp, int Kp) {
  int idx = blockIdx.x * blockDim.x + threadIdx.x;
  if (idx >= Mp * Kp) return;
  int r = idx / Kp, c = idx - r * Kp;
  float v = (r < M && c < K) ? in[(size_t)r * K + c] : 0.f;
  out[idx] = (_Float16)v;
}

// ============================================================
// im2col for 3x3 pad1 conv, output f16 [H*W, Kp], K = C*9
// chw==1: input [C,H,W];  chw==0: input [H*W, C] (channels-last)
// ============================================================
__global__ void im2col3x3_kernel(const float* __restrict__ in, _Float16* __restrict__ out,
                                 int C, int H, int W, int chw, int Kp) {
  int idx = blockIdx.x * blockDim.x + threadIdx.x;
  if (idx >= H * W * Kp) return;
  int m = idx / Kp, k = idx - m * Kp;
  float v = 0.f;
  if (k < C * 9) {
    int c = k / 9, t = k - c * 9;
    int ky = t / 3, kx = t - ky * 3;
    int y = m / W, x = m - y * W;
    int yy = y + ky - 1, xx = x + kx - 1;
    if (yy >= 0 && yy < H && xx >= 0 && xx < W)
      v = chw ? in[((size_t)c * H + yy) * W + xx]
              : in[((size_t)yy * W + xx) * C + c];
  }
  out[idx] = (_Float16)v;
}

// ============================================================
// WMMA GEMM: C[M,N] = act(scale * A[M,K] * B[N,K]^T + bias[N])
// A f16 [ceil(M/32)*32, Kp] row-major zero-padded, B f16 [ceil(N/32)*32, Kp].
// One wave computes a 32x32 C tile: 2 A-frags x 2 B-frags -> 4 wmma per
// 32-wide K step (128 B of fragment loads per 4 x v_wmma_f32_16x16x32_f16).
// 4 waves per block, one tile each. Fragment k order: contiguous, k = 16*hi + s, same for A and B.
// ============================================================
__device__ __forceinline__ float apply_act(float v, int act) {
  if      (act == ACT_RELU)  v = fmaxf(v, 0.f);
  else if (act == ACT_LRELU) v = (v > 0.f) ? v : 0.1f * v;
  else if (act == ACT_TANH)  v = tanhf(v);
  else if (act == ACT_SIG)   v = 1.f / (1.f + expf(-v));
  return v;
}

__global__ __launch_bounds__(128)
void wmma_gemm_kernel(const _Float16* __restrict__ A,
                      const _Float16* __restrict__ B,
                      const float* __restrict__ bias,
                      float* __restrict__ C,
                      int M, int N, int Kp, float scale, int act,
                      int tilesN, int totalTiles) {
  int tile = blockIdx.x * (blockDim.x >> 5) + (threadIdx.x >> 5);
  if (tile >= totalTiles) return;
  int tm = tile / tilesN, tn = tile - tm * tilesN;
  int lane = threadIdx.x & 31;
  int r  = lane & 15;
  int hi = lane >> 4;            // 0: low K-half group, 1: high
  const _Float16* a0p = A + (size_t)(tm * 32 + r) * Kp + hi * 16;
  const _Float16* a1p = a0p + (size_t)16 * Kp;
  const _Float16* b0p = B + (size_t)(tn * 32 + r) * Kp + hi * 16;
  const _Float16* b1p = b0p + (size_t)16 * Kp;

  v8f acc00 = {}, acc01 = {}, acc10 = {}, acc11 = {};
  for (int k = 0; k < Kp; k += 32) {
    __builtin_prefetch(a0p + k + 32, 0, 1);   // global_prefetch_b8
    __builtin_prefetch(b0p + k + 32, 0, 1);
    V16H a0, a1, b0, b1;
    a0.h[0] = *(const v8h*)(a0p + k);
    a0.h[1] = *(const v8h*)(a0p + k + 8);
    a1.h[0] = *(const v8h*)(a1p + k);
    a1.h[1] = *(const v8h*)(a1p + k + 8);
    b0.v    = *(const v16h*)(b0p + k);
    b1.v    = *(const v16h*)(b1p + k);
    acc00 = __builtin_amdgcn_wmma_f32_16x16x32_f16(false, a0.v, false, b0.v, (short)0, acc00, false, false);
    acc01 = __builtin_amdgcn_wmma_f32_16x16x32_f16(false, a0.v, false, b1.v, (short)0, acc01, false, false);
    acc10 = __builtin_amdgcn_wmma_f32_16x16x32_f16(false, a1.v, false, b0.v, (short)0, acc10, false, false);
    acc11 = __builtin_amdgcn_wmma_f32_16x16x32_f16(false, a1.v, false, b1.v, (short)0, acc11, false, false);
  }

  v8f accs[4] = {acc00, acc01, acc10, acc11};
  for (int q = 0; q < 4; ++q) {
    int mb = tm * 32 + (q >> 1) * 16;
    int nb = tn * 32 + (q & 1) * 16;
    int n = nb + r;
    if (n < N) {
      float bv = bias ? bias[n] : 0.f;
      int mbase = mb + hi * 8;
      for (int j = 0; j < 8; ++j) {
        int m = mbase + j;
        if (m < M)
          C[(size_t)m * N + n] = apply_act(accs[q][j] * scale + bv, act);
      }
    }
  }
}

// ============================================================
// pixel_shuffle(2) + leaky_relu(0.1): [576,1024] (pix-major) -> [2304,256] HWC
// ============================================================
__global__ void pixshuf_kernel(const float* __restrict__ c0, float* __restrict__ xs) {
  int idx = blockIdx.x * blockDim.x + threadIdx.x;
  if (idx >= HIDC * NPIX) return;
  int c = idx / NPIX, p = idx - c * NPIX;
  int oy = p / WF, ox = p - oy * WF;
  int h = oy >> 1, i = oy & 1, w = ox >> 1, j = ox & 1;
  float v = c0[((size_t)h * W_IN + w) * C4 + (c * 4 + i * 2 + j)];
  v = (v > 0.f) ? v : 0.1f * v;
  xs[(size_t)p * HIDC + c] = v;
}

// ============================================================
// Gaussian dictionary: 9x9x9 grid + zero row -> gd f32 [730,3], gdh f16 [736,32]
// ============================================================
__global__ void gaudict_kernel(float* __restrict__ gd, _Float16* __restrict__ gdh) {
  int n = blockIdx.x * blockDim.x + threadIdx.x;
  if (n >= NDP) return;
  float v0 = 0.f, v1 = 0.f, v2 = 0.f;
  if (n < ND - 1) {
    int i = n / 81, j = (n / 9) % 9, k = n % 9;
    v0 = 2.5f * i / 8.f;
    v1 = -1.f + 2.5f * j / 8.f;
    v2 = 2.5f * k / 8.f;
  }
  if (n < ND) { gd[n * 3] = v0; gd[n * 3 + 1] = v1; gd[n * 3 + 2] = v2; }
  for (int t = 0; t < 32; ++t) gdh[(size_t)n * 32 + t] = (_Float16)0.f;
  if (n < ND) {
    gdh[(size_t)n * 32 + 0] = (_Float16)v0;
    gdh[(size_t)n * 32 + 1] = (_Float16)v1;
    gdh[(size_t)n * 32 + 2] = (_Float16)v2;
  }
}

// ============================================================
// fused softmax + cov = softmax(scores) @ gd ; one block per row of [2304,730]
// ============================================================
__global__ void softmax_cov_kernel(const float* __restrict__ scores,
                                   const float* __restrict__ gd,
                                   float* __restrict__ cov) {
  __shared__ float red[256];
  int m = blockIdx.x, tid = threadIdx.x;
  const float* s = scores + (size_t)m * ND;

  float mx = -1e30f;
  for (int n = tid; n < ND; n += 256) mx = fmaxf(mx, s[n]);
  red[tid] = mx; __syncthreads();
  for (int st = 128; st > 0; st >>= 1) {
    if (tid < st) red[tid] = fmaxf(red[tid], red[tid + st]);
    __syncthreads();
  }
  mx = red[0]; __syncthreads();

  float ps = 0.f, p0 = 0.f, p1 = 0.f, p2 = 0.f;
  for (int n = tid; n < ND; n += 256) {
    float e = expf(s[n] - mx);
    ps += e;
    p0 += e * gd[n * 3];
    p1 += e * gd[n * 3 + 1];
    p2 += e * gd[n * 3 + 2];
  }
  float vals[4] = {ps, p0, p1, p2};
  float res[4];
  for (int q = 0; q < 4; ++q) {
    red[tid] = vals[q]; __syncthreads();
    for (int st = 128; st > 0; st >>= 1) {
      if (tid < st) red[tid] += red[tid + st];
      __syncthreads();
    }
    res[q] = red[0]; __syncthreads();
  }
  if (tid == 0) {
    float inv = 1.f / res[0];
    cov[m * 3 + 0] = res[1] * inv;
    cov[m * 3 + 1] = res[2] * inv;
    cov[m * 3 + 2] = res[3] * inv;
  }
}

// ============================================================
// per-gaussian params: coords, inverse covariance, opacity, colors -> [2304,9]
// ============================================================
__global__ void params_kernel(const float* __restrict__ cov, const float* __restrict__ off,
                              const float* __restrict__ opac, const float* __restrict__ colors,
                              float* __restrict__ prm) {
  int m = blockIdx.x * blockDim.x + threadIdx.x;
  if (m >= NPIX) return;
  const float step = 2.f / 48.f;
  float cy = -1.f + 1.f / 48.f + step * (float)(m / WF) + off[m * 2 + 0] * step;
  float cx = -1.f + 1.f / 48.f + step * (float)(m % WF) + off[m * 2 + 1] * step;
  float c1 = cov[m * 3 + 0] * 2.f + 1e-4f;   // scale = H_out/H = 2
  float c2 = cov[m * 3 + 1] * 2.f;
  float c3 = cov[m * 3 + 2] * 2.f + 1e-4f;
  float sxx = c1 * c1, sxy = c1 * c2, syy = c2 * c2 + c3 * c3;
  float det = sxx * syy - sxy * sxy + 1e-6f;
  float inv = 1.f / det;
  float* p = prm + (size_t)m * 9;
  p[0] = cy; p[1] = cx;
  p[2] = syy * inv; p[3] = -sxy * inv; p[4] = sxx * inv;
  p[5] = opac[m];
  p[6] = colors[m * 3]; p[7] = colors[m * 3 + 1]; p[8] = colors[m * 3 + 2];
}

// ============================================================
// bicubic (a=-0.75, align_corners=False, border clamp) 24 -> 96 both axes
// ============================================================
__device__ __forceinline__ float cubw(float t) {
  const float a = -0.75f;
  float at = fabsf(t);
  if (at <= 1.f) return ((a + 2.f) * at - (a + 3.f)) * at * at + 1.f;
  if (at < 2.f)  return ((a * at - 5.f * a) * at + 8.f * a) * at - 4.f * a;
  return 0.f;
}
__global__ void bicubic_kernel(const float* __restrict__ lr, float* __restrict__ up) {
  int idx = blockIdx.x * blockDim.x + threadIdx.x;
  if (idx >= 3 * PIXO) return;
  int c = idx / PIXO, p = idx - c * PIXO;
  int ho = p / WO, wo = p - ho * WO;
  float py = ((float)ho + 0.5f) * 0.25f - 0.5f;
  float px = ((float)wo + 0.5f) * 0.25f - 0.5f;
  int iy = (int)floorf(py), ix = (int)floorf(px);
  float acc = 0.f;
  for (int ty = -1; ty <= 2; ++ty) {
    int yc = min(max(iy + ty, 0), 23);
    float wy = cubw(py - (float)(iy + ty));
    for (int tx = -1; tx <= 2; ++tx) {
      int xc = min(max(ix + tx, 0), 23);
      float wx = cubw(px - (float)(ix + tx));
      acc += wy * wx * lr[(size_t)c * (H_IN * W_IN) + yc * W_IN + xc];
    }
  }
  up[idx] = acc;
}

// ============================================================
// dense splat: each thread = one output pixel; gaussians staged via LDS
// ============================================================
__global__ void splat_kernel(const float* __restrict__ prm, const float* __restrict__ up,
                             float* __restrict__ out) {
  __shared__ float lds[256 * 9];
  int tid = threadIdx.x;
  int pix = blockIdx.x * 256 + tid;
  int ho = pix / WO, wo = pix - ho * WO;
  float yy = -1.f + 2.f * (float)ho / 95.f;
  float xx = -1.f + 2.f * (float)wo / 95.f;
  float ws = 0.f, ar = 0.f, ag = 0.f, ab = 0.f;
  for (int g0 = 0; g0 < NPIX; g0 += 256) {
    __syncthreads();
    for (int i = tid; i < 256 * 9; i += 256) lds[i] = prm[(size_t)g0 * 9 + i];
    __syncthreads();
    for (int j = 0; j < 256; ++j) {
      const float* p = &lds[j * 9];
      float dx = yy - p[0], dy = xx - p[1];
      float mah = dx * dx * p[2] + 2.f * dx * dy * p[3] + dy * dy * p[4];
      float w = expf(-0.5f * mah) * p[5];
      ws += w; ar += w * p[6]; ag += w * p[7]; ab += w * p[8];
    }
  }
  float inv = 1.f / (ws + 1e-6f);
  float alpha = 1.f - expf(-ws);
  out[pix]            = ar * inv * alpha + (1.f - alpha) + up[pix];
  out[PIXO + pix]     = ag * inv * alpha + (1.f - alpha) + up[PIXO + pix];
  out[2 * PIXO + pix] = ab * inv * alpha + (1.f - alpha) + up[2 * PIXO + pix];
}

// ============================================================
// host orchestration
// ============================================================
extern "C" void kernel_launch(void* const* d_in, const int* in_sizes, int n_in,
                              void* d_out, int out_size, void* d_ws, size_t ws_size,
                              hipStream_t stream) {
  (void)in_sizes; (void)n_in; (void)out_size; (void)ws_size;
  const float* feat = (const float*)d_in[0];
  const float* lr   = (const float*)d_in[1];
  const float* fc0w = (const float*)d_in[2];  const float* fc0b = (const float*)d_in[3];
  const float* fc1w = (const float*)d_in[4];  const float* fc1b = (const float*)d_in[5];
  const float* cm0w = (const float*)d_in[6];  const float* cm0b = (const float*)d_in[7];
  const float* cm1w = (const float*)d_in[8];  const float* cm1b = (const float*)d_in[9];
  const float* cm2w = (const float*)d_in[10]; const float* cm2b = (const float*)d_in[11];
  const float* om0w = (const float*)d_in[12]; const float* om0b = (const float*)d_in[13];
  const float* om1w = (const float*)d_in[14]; const float* om1b = (const float*)d_in[15];
  const float* cs0w = (const float*)d_in[16]; const float* cs0b = (const float*)d_in[17];
  const float* cs1w = (const float*)d_in[18]; const float* cs1b = (const float*)d_in[19];
  const float* de0w = (const float*)d_in[20]; const float* de0b = (const float*)d_in[21];
  const float* de1w = (const float*)d_in[22]; const float* de1b = (const float*)d_in[23];
  const float* de2w = (const float*)d_in[24]; const float* de2b = (const float*)d_in[25];
  const float* op0w = (const float*)d_in[26]; const float* op0b = (const float*)d_in[27];
  const float* op1w = (const float*)d_in[28]; const float* op1b = (const float*)d_in[29];

  size_t cur = 0;
  auto alloc = [&](size_t bytes) -> char* {
    cur = (cur + 255) & ~(size_t)255;
    char* p = (char*)d_ws + cur;
    cur += bytes;
    return p;
  };
  // f16 scratch (B matrices padded to multiples of 32 rows for 32-wide N tiles)
  _Float16* A0    = (_Float16*)alloc((size_t)576 * 576 * 2);
  _Float16* A1    = (_Float16*)alloc((size_t)NPIX * 2304 * 2);
  _Float16* Wfc0  = (_Float16*)alloc((size_t)1024 * 576 * 2);
  _Float16* Wfc1  = (_Float16*)alloc((size_t)256 * 2304 * 2);
  _Float16* Wcm0  = (_Float16*)alloc((size_t)256 * 256 * 2);
  _Float16* Wcm1  = (_Float16*)alloc((size_t)128 * 256 * 2);
  _Float16* Wcm2  = (_Float16*)alloc((size_t)32 * 128 * 2);
  _Float16* Wom0  = (_Float16*)alloc((size_t)128 * 256 * 2);
  _Float16* Wom1  = (_Float16*)alloc((size_t)32 * 128 * 2);
  _Float16* Wcs0  = (_Float16*)alloc((size_t)256 * 2304 * 2);
  _Float16* Wcs1  = (_Float16*)alloc((size_t)512 * 256 * 2);
  _Float16* Wde0  = (_Float16*)alloc((size_t)128 * 32 * 2);
  _Float16* Wde1  = (_Float16*)alloc((size_t)256 * 128 * 2);
  _Float16* Wde2  = (_Float16*)alloc((size_t)512 * 256 * 2);
  _Float16* Wop0  = (_Float16*)alloc((size_t)64 * 256 * 2);
  _Float16* Wop1  = (_Float16*)alloc((size_t)32 * 64 * 2);
  _Float16* Xf16  = (_Float16*)alloc((size_t)NPIX * 256 * 2);
  _Float16* H1h   = (_Float16*)alloc((size_t)NPIX * 256 * 2);
  _Float16* H2h   = (_Float16*)alloc((size_t)NPIX * 128 * 2);
  _Float16* T16   = (_Float16*)alloc((size_t)NPIX * 256 * 2);
  _Float16* SEL16 = (_Float16*)alloc((size_t)NPIX * 512 * 2);
  _Float16* GD16  = (_Float16*)alloc((size_t)NDP * 32 * 2);
  _Float16* D0h   = (_Float16*)alloc((size_t)NDP * 128 * 2);
  _Float16* D1h   = (_Float16*)alloc((size_t)NDP * 256 * 2);
  _Float16* DE16  = (_Float16*)alloc((size_t)NDP * 512 * 2);
  // f32 scratch
  float* Cfc0   = (float*)alloc((size_t)576 * 1024 * 4);
  float* XS     = (float*)alloc((size_t)NPIX * 256 * 4);
  float* X      = (float*)alloc((size_t)NPIX * 256 * 4);
  float* H1     = (float*)alloc((size_t)NPIX * 256 * 4);
  float* H2     = (float*)alloc((size_t)NPIX * 128 * 4);
  float* COLORS = (float*)alloc((size_t)NPIX * 3 * 4);
  float* OFF    = (float*)alloc((size_t)NPIX * 2 * 4);
  float* OPAC   = (float*)alloc((size_t)NPIX * 4);
  float* T      = (float*)alloc((size_t)NPIX * 256 * 4);
  float* SEL    = (float*)alloc((size_t)NPIX * 512 * 4);
  float* GD     = (float*)alloc((size_t)ND * 3 * 4);
  float* D0     = (float*)alloc((size_t)ND * 128 * 4);
  float* D1     = (float*)alloc((size_t)ND * 256 * 4);
  float* DE     = (float*)alloc((size_t)ND * 512 * 4);
  float* SCORES = (float*)alloc((size_t)NPIX * ND * 4);
  float* COV    = (float*)alloc((size_t)NPIX * 3 * 4);
  float* PRM    = (float*)alloc((size_t)NPIX * 9 * 4);
  float* LRUP   = (float*)alloc((size_t)3 * PIXO * 4);

  auto blocks = [](long total) { return dim3((unsigned)((total + 255) / 256)); };
  auto pack = [&](const float* in, _Float16* out, int M, int K, int Mp, int Kp) {
    pack_f16_kernel<<<blocks((long)Mp * Kp), 256, 0, stream>>>(in, out, M, K, Mp, Kp);
  };
  auto gemm = [&](const _Float16* A, const _Float16* B, const float* bias, float* C,
                  int M, int N, int Kp, float scale, int act) {
    int tm = (M + 31) / 32, tn = (N + 31) / 32;
    int total = tm * tn;
    wmma_gemm_kernel<<<dim3((total + 3) / 4), dim3(128), 0, stream>>>(
        A, B, bias, C, M, N, Kp, scale, act, tn, total);
  };

  // weights -> f16 (conv [O,I,3,3] flattens directly to [N, K=I*9])
  pack(fc0w, Wfc0, 1024, 576, 1024, 576);
  pack(fc1w, Wfc1, 256, 2304, 256, 2304);
  pack(cm0w, Wcm0, 256, 256, 256, 256);
  pack(cm1w, Wcm1, 128, 256, 128, 256);
  pack(cm2w, Wcm2, 3, 128, 32, 128);
  pack(om0w, Wom0, 128, 256, 128, 256);
  pack(om1w, Wom1, 2, 128, 32, 128);
  pack(cs0w, Wcs0, 256, 2304, 256, 2304);
  pack(cs1w, Wcs1, 512, 256, 512, 256);
  pack(de0w, Wde0, 128, 3, 128, 32);
  pack(de1w, Wde1, 256, 128, 256, 128);
  pack(de2w, Wde2, 512, 256, 512, 256);
  pack(op0w, Wop0, 64, 256, 64, 256);
  pack(op1w, Wop1, 1, 64, 32, 64);

  // feat_conv: conv3x3 64->1024 on 24x24
  im2col3x3_kernel<<<blocks((long)576 * 576), 256, 0, stream>>>(feat, A0, C_IN, H_IN, W_IN, 1, 576);
  gemm(A0, Wfc0, fc0b, Cfc0, 576, 1024, 576, 1.f, ACT_NONE);
  // pixel_shuffle(2) + leaky_relu
  pixshuf_kernel<<<blocks((long)HIDC * NPIX), 256, 0, stream>>>(Cfc0, XS);
  // conv3x3 256->256 on 48x48 (fc1)
  im2col3x3_kernel<<<blocks((long)NPIX * 2304), 256, 0, stream>>>(XS, A1, HIDC, HF, WF, 0, 2304);
  gemm(A1, Wfc1, fc1b, X, NPIX, 256, 2304, 1.f, ACT_NONE);
  pack(X, Xf16, NPIX, 256, NPIX, 256);

  // color MLP: 256 -> 256 (relu) -> 128 (relu) -> 3
  gemm(Xf16, Wcm0, cm0b, H1, NPIX, 256, 256, 1.f, ACT_RELU);
  pack(H1, H1h, NPIX, 256, NPIX, 256);
  gemm(H1h, Wcm1, cm1b, H2, NPIX, 128, 256, 1.f, ACT_RELU);
  pack(H2, H2h, NPIX, 128, NPIX, 128);
  gemm(H2h, Wcm2, cm2b, COLORS, NPIX, 3, 128, 1.f, ACT_NONE);

  // offset MLP: 256 -> 128 (relu) -> 2 (tanh)
  gemm(Xf16, Wom0, om0b, H2, NPIX, 128, 256, 1.f, ACT_RELU);
  pack(H2, H2h, NPIX, 128, NPIX, 128);
  gemm(H2h, Wom1, om1b, OFF, NPIX, 2, 128, 1.f, ACT_TANH);

  // opacity MLP: 256 -> 64 (relu) -> 1 (sigmoid)
  gemm(Xf16, Wop0, op0b, H2, NPIX, 64, 256, 1.f, ACT_RELU);
  pack(H2, H2h, NPIX, 64, NPIX, 64);
  gemm(H2h, Wop1, op1b, OPAC, NPIX, 1, 64, 1.f, ACT_SIG);

  // cov selector: conv3x3 (lrelu) -> conv1x1 256->512
  im2col3x3_kernel<<<blocks((long)NPIX * 2304), 256, 0, stream>>>(X, A1, HIDC, HF, WF, 0, 2304);
  gemm(A1, Wcs0, cs0b, T, NPIX, 256, 2304, 1.f, ACT_LRELU);
  pack(T, T16, NPIX, 256, NPIX, 256);
  gemm(T16, Wcs1, cs1b, SEL, NPIX, 512, 256, 1.f, ACT_NONE);
  pack(SEL, SEL16, NPIX, 512, NPIX, 512);

  // dict embed: gd [730,3] -> 128 (relu) -> 256 (relu) -> 512
  gaudict_kernel<<<blocks(NDP), 256, 0, stream>>>(GD, GD16);
  gemm(GD16, Wde0, de0b, D0, ND, 128, 32, 1.f, ACT_RELU);
  pack(D0, D0h, ND, 128, NDP, 128);
  gemm(D0h, Wde1, de1b, D1, ND, 256, 128, 1.f, ACT_RELU);
  pack(D1, D1h, ND, 256, NDP, 256);
  gemm(D1h, Wde2, de2b, DE, ND, 512, 256, 1.f, ACT_NONE);
  pack(DE, DE16, ND, 512, NDP, 512);

  // scores = (sel @ de.T) / sqrt(512); softmax + cov fused (wts never stored)
  gemm(SEL16, DE16, nullptr, SCORES, NPIX, ND, 512, 0.044194173824159216f, ACT_NONE);
  softmax_cov_kernel<<<dim3(NPIX), 256, 0, stream>>>(SCORES, GD, COV);

  // splat + bicubic residual
  params_kernel<<<blocks(NPIX), 256, 0, stream>>>(COV, OFF, OPAC, COLORS, PRM);
  bicubic_kernel<<<blocks((long)3 * PIXO), 256, 0, stream>>>(lr, LRUP);
  splat_kernel<<<dim3(PIXO / 256), 256, 0, stream>>>(PRM, LRUP, (float*)d_out);
}
